// SparseMoELayer_48576080118265
// MI455X (gfx1250) — compile-verified, matrix operand order checked by reading, not run on hardware
//
#include <hip/hip_runtime.h>
#include <hip/hip_bf16.h>
#include <math.h>

typedef __attribute__((ext_vector_type(16))) __bf16 v16bf;
typedef __attribute__((ext_vector_type(8)))  __bf16 v8bf;
typedef __attribute__((ext_vector_type(8)))  float  v8f;

namespace {
constexpr int kD   = 512;
constexpr int kE   = 8;
constexpr int kK   = 2;
constexpr int kF   = 2048;
constexpr int kN   = 4 * 8192;        // 32768 tokens
constexpr int kCAP = 5120;            // ceil(1.25 * N / E)
constexpr int kENT = kN * kK;         // 65536 routed entries
constexpr int kHB  = 1024;            // entries per routing block
constexpr int kNHB = kENT / kHB;      // 64 routing blocks
constexpr int kLDSPITCH = 40;         // 32 bf16 + 8 pad -> 80B row, bank-spread
}

// ---------------------------------------------------------------------------
// CDNA5 async global->LDS copy (16B per lane, ASYNCcnt-tracked, no VGPR data).
// VDST operand carries the LDS byte address; flat shared pointers map to LDS
// via addr[31:0] (ISA 10.2 aperture rules), so truncation gives the offset.
// ---------------------------------------------------------------------------
__device__ inline void async_cp16(void* lds_dst, const void* gsrc) {
  unsigned loff = (unsigned)(size_t)lds_dst;
  unsigned long long ga = (unsigned long long)(size_t)gsrc;
  asm volatile("global_load_async_to_lds_b128 %0, %1, off"
               :: "v"(loff), "v"(ga) : "memory");
}

__device__ inline void wait_async0() {
#if __has_builtin(__builtin_amdgcn_s_wait_asynccnt)
  __builtin_amdgcn_s_wait_asynccnt(0);
#else
  asm volatile("s_wait_asynccnt 0" ::: "memory");
#endif
}

// ---------------------------------------------------------------------------
// 1) Gating: one wave per token. logits = x @ gate_w^T, softmax, top-2,
//    renormalized weights. Deterministic tie-break = lowest index (strict >).
// ---------------------------------------------------------------------------
__global__ void k_gate(const float* __restrict__ x, const float* __restrict__ gw,
                       int* __restrict__ tidx, float* __restrict__ tw) {
  const int wave = (blockIdx.x * blockDim.x + threadIdx.x) >> 5;
  const int lane = threadIdx.x & 31;
  if (wave >= kN) return;
  const float4* xv = (const float4*)(x + (size_t)wave * kD + lane * 16);
  float4 xr[4];
#pragma unroll
  for (int q = 0; q < 4; ++q) xr[q] = xv[q];
  float acc[kE];
#pragma unroll
  for (int e = 0; e < kE; ++e) {
    const float4* gv = (const float4*)(gw + e * kD + lane * 16);
    float s = 0.f;
#pragma unroll
    for (int q = 0; q < 4; ++q) {
      float4 g = gv[q];
      s += xr[q].x * g.x + xr[q].y * g.y + xr[q].z * g.z + xr[q].w * g.w;
    }
    acc[e] = s;
  }
#pragma unroll
  for (int e = 0; e < kE; ++e)
#pragma unroll
    for (int off = 16; off > 0; off >>= 1)
      acc[e] += __shfl_xor(acc[e], off, 32);
  if (lane == 0) {
    float mx = acc[0];
#pragma unroll
    for (int e = 1; e < kE; ++e) mx = fmaxf(mx, acc[e]);
    float p[kE], s = 0.f;
#pragma unroll
    for (int e = 0; e < kE; ++e) { p[e] = __expf(acc[e] - mx); s += p[e]; }
    const float inv = 1.f / s;
#pragma unroll
    for (int e = 0; e < kE; ++e) p[e] *= inv;
    int i0 = 0;
#pragma unroll
    for (int e = 1; e < kE; ++e) if (p[e] > p[i0]) i0 = e;
    int i1 = (i0 == 0) ? 1 : 0;
#pragma unroll
    for (int e = 0; e < kE; ++e) if (e != i0 && p[e] > p[i1]) i1 = e;
    const float s2 = 1.f / (p[i0] + p[i1]);
    tidx[wave * 2 + 0] = i0;  tidx[wave * 2 + 1] = i1;
    tw[wave * 2 + 0] = p[i0] * s2;  tw[wave * 2 + 1] = p[i1] * s2;
  }
}

// ---------------------------------------------------------------------------
// 2a) Per-block expert histogram over flat entries (reference cumsum order).
// ---------------------------------------------------------------------------
__global__ void k_hist(const int* __restrict__ tidx, int* __restrict__ bcount) {
  __shared__ int cnt[kE];
  if (threadIdx.x < kE) cnt[threadIdx.x] = 0;
  __syncthreads();
  const int i = blockIdx.x * kHB + threadIdx.x;
  if (i < kENT) atomicAdd(&cnt[tidx[i]], 1);
  __syncthreads();
  if (threadIdx.x < kE) bcount[blockIdx.x * kE + threadIdx.x] = cnt[threadIdx.x];
}

// 2b) Exclusive scan of block counts per expert (tiny; 8 lanes).
__global__ void k_scan(const int* __restrict__ bcount, int* __restrict__ boff) {
  const int e = threadIdx.x;
  if (e >= kE) return;
  int run = 0;
  for (int b = 0; b < kNHB; ++b) { boff[b * kE + e] = run; run += bcount[b * kE + e]; }
}

// 2c) Final per-entry position: block offset + wave prefix + in-wave rank.
__global__ void k_assign(const int* __restrict__ tidx, const int* __restrict__ boff,
                         int* __restrict__ epos) {
  __shared__ int wcnt[kHB / 32][kE];
  const int i    = blockIdx.x * kHB + threadIdx.x;
  const int lane = threadIdx.x & 31;
  const int wid  = threadIdx.x >> 5;
  const int e    = tidx[i];
  unsigned my_mask = 0u;
#pragma unroll
  for (int q = 0; q < kE; ++q) {
    const unsigned m = __builtin_amdgcn_ballot_w32(e == q);
    if (q == e) my_mask = m;
    if (lane == q) wcnt[wid][q] = __popc(m);
  }
  __syncthreads();
  const unsigned ltm = (lane == 0) ? 0u : ((1u << lane) - 1u);
  int pos = __popc(my_mask & ltm) + boff[blockIdx.x * kE + e];
  for (int w = 0; w < wid; ++w) pos += wcnt[w][e];
  epos[i] = pos;
}

// ---------------------------------------------------------------------------
// 3) Batched transpose + f32->bf16 convert: out[e][c][r] = (bf16) in[e][r][c]
// ---------------------------------------------------------------------------
__global__ void k_transpose_cvt(const float* __restrict__ in, __bf16* __restrict__ out,
                                int R, int C) {
  __shared__ float tile[32][33];
  const float* ip = in  + (size_t)blockIdx.z * R * C;
  __bf16*      op = out + (size_t)blockIdx.z * R * C;
  const int c0 = blockIdx.x * 32, r0 = blockIdx.y * 32;
  tile[threadIdx.y][threadIdx.x] = ip[(size_t)(r0 + threadIdx.y) * C + c0 + threadIdx.x];
  __syncthreads();
  op[(size_t)(c0 + threadIdx.y) * R + r0 + threadIdx.x] =
      (__bf16)tile[threadIdx.x][threadIdx.y];
}

// ---------------------------------------------------------------------------
// 4) Gather kept tokens into expert buffers (E, CAP, D) bf16. One wave/entry.
// ---------------------------------------------------------------------------
__global__ void k_gather(const float* __restrict__ x, const int* __restrict__ tidx,
                         const int* __restrict__ epos, __bf16* __restrict__ xbuf) {
  const int ent  = (blockIdx.x * blockDim.x + threadIdx.x) >> 5;
  const int lane = threadIdx.x & 31;
  if (ent >= kENT) return;
  const int pos = epos[ent];
  if (pos >= kCAP) return;                 // dropped (over capacity)
  const int e = tidx[ent];
  const int token = ent >> 1;
  const float4* src = (const float4*)(x + (size_t)token * kD + lane * 16);
  __bf16* dst = xbuf + ((size_t)e * kCAP + pos) * kD + lane * 16;
  float v[16];
#pragma unroll
  for (int q = 0; q < 4; ++q) {
    float4 t = src[q];
    v[4 * q + 0] = t.x; v[4 * q + 1] = t.y; v[4 * q + 2] = t.z; v[4 * q + 3] = t.w;
  }
  union { v8bf vec; __bf16 s[8]; } u0, u1;
#pragma unroll
  for (int q = 0; q < 8; ++q) { u0.s[q] = (__bf16)v[q]; u1.s[q] = (__bf16)v[8 + q]; }
  *(v8bf*)(dst)     = u0.vec;
  *(v8bf*)(dst + 8) = u1.vec;
}

// ---------------------------------------------------------------------------
// WMMA 16x32 bf16 fragment from an LDS tile row block (pitch kLDSPITCH).
// ISA layout: lanes 0-15 -> row, K 0-7 in v0-3 / 16-23 in v4-7;
//             lanes 16-31 -> same rows, K 8-15 / 24-31. (A and B identical.)
// ---------------------------------------------------------------------------
__device__ inline v16bf frag_lds(const __bf16* __restrict__ tile_base, int lane) {
  const int r  = lane & 15;
  const int kb = (lane >> 4) * 8;
  const __bf16* p = tile_base + r * kLDSPITCH + kb;
  union { v16bf v; v8bf h[2]; } u;
  u.h[0] = *(const v8bf*)(p);
  u.h[1] = *(const v8bf*)(p + 16);
  return u.v;
}

// ---------------------------------------------------------------------------
// Stage one BKx(128 A rows + 256 B rows) tile pair into LDS with async copies.
// 1536 x 16B chunks, 6 per thread (2 A + 4 B).
// ---------------------------------------------------------------------------
__device__ inline void stage_tiles(__bf16 (*sA)[kLDSPITCH], __bf16 (*sB)[kLDSPITCH],
                                   const __bf16* __restrict__ A, int ldA, int rowA0,
                                   const __bf16* __restrict__ Bt, int ldB, int colB0,
                                   int k, int tid) {
#pragma unroll
  for (int q = 0; q < 2; ++q) {
    const int ch = tid + 256 * q;
    const int row = ch >> 2, qq = ch & 3;
    async_cp16(&sA[row][qq * 8], A + (size_t)(rowA0 + row) * ldA + k + qq * 8);
  }
#pragma unroll
  for (int q = 0; q < 4; ++q) {
    const int ch = tid + 256 * q;
    const int row = ch >> 2, qq = ch & 3;
    async_cp16(&sB[row][qq * 8], Bt + (size_t)(colB0 + row) * ldB + k + qq * 8);
  }
}

// ---------------------------------------------------------------------------
// 5) GEMM1: h = gelu(xbuf @ w1t^T + b1), bf16 out. Block 256 thr = 8 waves
//    (2 M x 4 N), wave tile 64x64 (4x4 accum tiles). Double-buffered LDS
//    staging via global_load_async_to_lds_b128 + s_wait_asynccnt.
// ---------------------------------------------------------------------------
__global__ void __launch_bounds__(256)
k_gemm1(const __bf16* __restrict__ xbuf, const __bf16* __restrict__ w1t,
        const float* __restrict__ b1, __bf16* __restrict__ hbuf) {
  __shared__ __attribute__((aligned(16))) __bf16 sA[2][128][kLDSPITCH];
  __shared__ __attribute__((aligned(16))) __bf16 sB[2][256][kLDSPITCH];
  const int e    = blockIdx.z;
  const int tid  = threadIdx.x;
  const int lane = tid & 31;
  const int wv   = tid >> 5;
  const int rowA0 = blockIdx.y * 128;
  const int colB0 = blockIdx.x * 256;
  const int wrow = (wv & 1) * 64;     // wave row offset inside block tile
  const int wcol = (wv >> 1) * 64;    // wave col offset inside block tile
  const __bf16* A  = xbuf + (size_t)e * kCAP * kD;
  const __bf16* Bt = w1t  + (size_t)e * kF   * kD;   // (F, D) bf16

  v8f acc[4][4] = {};
  stage_tiles(sA[0], sB[0], A, kD, rowA0, Bt, kD, colB0, 0, tid);
  wait_async0();
  __syncthreads();

  for (int k = 0; k < kD; k += 32) {
    const int cur = (k >> 5) & 1;
    const bool more = (k + 32) < kD;
    if (more)
      stage_tiles(sA[cur ^ 1], sB[cur ^ 1], A, kD, rowA0, Bt, kD, colB0, k + 32, tid);
    v16bf a[4], b[4];
#pragma unroll
    for (int i = 0; i < 4; ++i) a[i] = frag_lds(&sA[cur][wrow + i * 16][0], lane);
#pragma unroll
    for (int j = 0; j < 4; ++j) b[j] = frag_lds(&sB[cur][wcol + j * 16][0], lane);
#pragma unroll
    for (int i = 0; i < 4; ++i)
#pragma unroll
      for (int j = 0; j < 4; ++j)
        acc[i][j] = __builtin_amdgcn_wmma_f32_16x16x32_bf16(
            false, a[i], false, b[j], (short)0, acc[i][j], false, false);
    if (more) {
      wait_async0();
      __syncthreads();
    }
  }
  // Epilogue: bias + exact GELU -> bf16. C layout: VGPR t -> M=t+8*(lane>=16).
  const int mhi  = (lane >> 4) * 8;
  const int ncol = lane & 15;
#pragma unroll
  for (int j = 0; j < 4; ++j) {
    const int n = colB0 + wcol + j * 16 + ncol;
    const float bias = b1[e * kF + n];
#pragma unroll
    for (int i = 0; i < 4; ++i) {
#pragma unroll
      for (int t = 0; t < 8; ++t) {
        const int m = rowA0 + wrow + i * 16 + mhi + t;
        const float v = acc[i][j][t] + bias;
        const float g = 0.5f * v * (1.0f + erff(v * 0.70710678118654752f));
        hbuf[((size_t)e * kCAP + m) * kF + n] = (__bf16)g;
      }
    }
  }
}

// ---------------------------------------------------------------------------
// 6) GEMM2: expert_out = h @ w2t^T + b2, f32 out. Same pipelined tiling.
// ---------------------------------------------------------------------------
__global__ void __launch_bounds__(256)
k_gemm2(const __bf16* __restrict__ hbuf, const __bf16* __restrict__ w2t,
        const float* __restrict__ b2, float* __restrict__ obuf) {
  __shared__ __attribute__((aligned(16))) __bf16 sA[2][128][kLDSPITCH];
  __shared__ __attribute__((aligned(16))) __bf16 sB[2][256][kLDSPITCH];
  const int e    = blockIdx.z;
  const int tid  = threadIdx.x;
  const int lane = tid & 31;
  const int wv   = tid >> 5;
  const int rowA0 = blockIdx.y * 128;
  const int colB0 = blockIdx.x * 256;
  const int wrow = (wv & 1) * 64;
  const int wcol = (wv >> 1) * 64;
  const __bf16* A  = hbuf + (size_t)e * kCAP * kF;
  const __bf16* Bt = w2t  + (size_t)e * kD   * kF;   // (D, F) bf16

  v8f acc[4][4] = {};
  stage_tiles(sA[0], sB[0], A, kF, rowA0, Bt, kF, colB0, 0, tid);
  wait_async0();
  __syncthreads();

  for (int k = 0; k < kF; k += 32) {
    const int cur = (k >> 5) & 1;
    const bool more = (k + 32) < kF;
    if (more)
      stage_tiles(sA[cur ^ 1], sB[cur ^ 1], A, kF, rowA0, Bt, kF, colB0, k + 32, tid);
    v16bf a[4], b[4];
#pragma unroll
    for (int i = 0; i < 4; ++i) a[i] = frag_lds(&sA[cur][wrow + i * 16][0], lane);
#pragma unroll
    for (int j = 0; j < 4; ++j) b[j] = frag_lds(&sB[cur][wcol + j * 16][0], lane);
#pragma unroll
    for (int i = 0; i < 4; ++i)
#pragma unroll
      for (int j = 0; j < 4; ++j)
        acc[i][j] = __builtin_amdgcn_wmma_f32_16x16x32_bf16(
            false, a[i], false, b[j], (short)0, acc[i][j], false, false);
    if (more) {
      wait_async0();
      __syncthreads();
    }
  }
  const int mhi  = (lane >> 4) * 8;
  const int ncol = lane & 15;
#pragma unroll
  for (int j = 0; j < 4; ++j) {
    const int n = colB0 + wcol + j * 16 + ncol;
    const float bias = b2[e * kD + n];
#pragma unroll
    for (int i = 0; i < 4; ++i)
#pragma unroll
      for (int t = 0; t < 8; ++t) {
        const int m = rowA0 + wrow + i * 16 + mhi + t;
        obuf[((size_t)e * kCAP + m) * kD + n] = acc[i][j][t] + bias;
      }
  }
}

// ---------------------------------------------------------------------------
// 7) Combine: out[token] = sum_k w_k * expert_out[e_k][pos_k] (dropped -> 0).
// ---------------------------------------------------------------------------
__global__ void k_combine(const float* __restrict__ obuf, const int* __restrict__ tidx,
                          const float* __restrict__ tw, const int* __restrict__ epos,
                          float* __restrict__ out) {
  const int token = (blockIdx.x * blockDim.x + threadIdx.x) >> 5;
  const int lane  = threadIdx.x & 31;
  if (token >= kN) return;
  float4 acc[4];
#pragma unroll
  for (int q = 0; q < 4; ++q) acc[q] = make_float4(0.f, 0.f, 0.f, 0.f);
#pragma unroll
  for (int k = 0; k < kK; ++k) {
    const int i = token * kK + k;
    const int pos = epos[i];
    if (pos < kCAP) {
      const float w = tw[i];
      const int e = tidx[i];
      const float4* src = (const float4*)(obuf + ((size_t)e * kCAP + pos) * kD + lane * 16);
#pragma unroll
      for (int q = 0; q < 4; ++q) {
        float4 s = src[q];
        acc[q].x += w * s.x; acc[q].y += w * s.y;
        acc[q].z += w * s.z; acc[q].w += w * s.w;
      }
    }
  }
  float4* dst = (float4*)(out + (size_t)token * kD + lane * 16);
#pragma unroll
  for (int q = 0; q < 4; ++q) dst[q] = acc[q];
}

// ---------------------------------------------------------------------------
extern "C" void kernel_launch(void* const* d_in, const int* in_sizes, int n_in,
                              void* d_out, int out_size, void* d_ws, size_t ws_size,
                              hipStream_t stream) {
  (void)in_sizes; (void)n_in; (void)out_size; (void)ws_size;
  const float* x  = (const float*)d_in[0];
  const float* gw = (const float*)d_in[1];
  const float* w1 = (const float*)d_in[2];
  const float* b1 = (const float*)d_in[3];
  const float* w2 = (const float*)d_in[4];
  const float* b2 = (const float*)d_in[5];
  float* out = (float*)d_out;

  char* ws = (char*)d_ws;
  size_t off = 0;
  auto carve = [&](size_t bytes) -> void* {
    off = (off + 255) & ~(size_t)255;
    void* p = ws + off;
    off += bytes;
    return p;
  };
  __bf16* w1t  = (__bf16*)carve((size_t)kE * kD * kF * sizeof(__bf16));   // (E,F,D)
  __bf16* w2t  = (__bf16*)carve((size_t)kE * kD * kF * sizeof(__bf16));   // (E,D,F)
  __bf16* xbuf = (__bf16*)carve((size_t)kE * kCAP * kD * sizeof(__bf16)); // (E,C,D)
  __bf16* hbuf = (__bf16*)carve((size_t)kE * kCAP * kF * sizeof(__bf16)); // (E,C,F)
  float*  obuf = (float*) carve((size_t)kE * kCAP * kD * sizeof(float));  // (E,C,D)
  int*    tidx = (int*)   carve((size_t)kENT * sizeof(int));
  float*  twv  = (float*) carve((size_t)kENT * sizeof(float));
  int*    epos = (int*)   carve((size_t)kENT * sizeof(int));
  int*    bcnt = (int*)   carve((size_t)kNHB * kE * sizeof(int));
  int*    boff = (int*)   carve((size_t)kNHB * kE * sizeof(int));

  // 1) gating (one wave per token)
  k_gate<<<kN / 8, 256, 0, stream>>>(x, gw, tidx, twv);
  // 2) deterministic routing (reference flat-order cumsum semantics)
  k_hist  <<<kNHB, kHB, 0, stream>>>(tidx, bcnt);
  k_scan  <<<1, 32, 0, stream>>>(bcnt, boff);
  k_assign<<<kNHB, kHB, 0, stream>>>(tidx, boff, epos);
  // 3) weight convert + transpose (B fragments become K-contiguous)
  k_transpose_cvt<<<dim3(kF / 32, kD / 32, kE), dim3(32, 32), 0, stream>>>(w1, w1t, kD, kF);
  k_transpose_cvt<<<dim3(kD / 32, kF / 32, kE), dim3(32, 32), 0, stream>>>(w2, w2t, kF, kD);
  // 4) gather tokens to expert buffers (one wave per routed entry)
  k_gather<<<kENT / 8, 256, 0, stream>>>(x, tidx, epos, xbuf);
  // 5/6) expert FFN via bf16 WMMA, async-LDS double-buffered pipeline
  k_gemm1<<<dim3(kF / 256, kCAP / 128, kE), 256, 0, stream>>>(xbuf, w1t, b1, hbuf);
  k_gemm2<<<dim3(kD / 256, kCAP / 128, kE), 256, 0, stream>>>(hbuf, w2t, b2, obuf);
  // 7) weighted combine back to token order
  k_combine<<<kN / 8, 256, 0, stream>>>(obuf, tidx, twv, epos, out);
}
